// LabelAttention_11742440587867
// MI455X (gfx1250) — matrix operand order repaired and checked by reading, not compiled, not run on hardware
//
#include <hip/hip_runtime.h>
#include <math.h>

// ---------------------------------------------------------------------------
// LabelAttention for MI455X (gfx1250, wave32, WMMA).
//   B=4, S=2048, H=512, E=256, L=8921.
//   GEMMs: v_wmma_f32_16x16x32_bf16, double-buffered LDS,
//   async global->LDS staging (ASYNCcnt) for row-major A tiles.
// ---------------------------------------------------------------------------

#define B_  4
#define S_  2048
#define H_  512
#define E_  256
#define L_  8921
#define LP_ 8960          // L padded to 70 * 128
#define NEG_INF_ (-1.0e9f)

typedef __attribute__((ext_vector_type(16))) __bf16 v16bf;
typedef __attribute__((ext_vector_type(8)))  float  v8f;

struct alignas(16) U128 { unsigned x, y, z, w; };
struct alignas(8)  U64  { unsigned x, y; };

union FragAB { U128 u[2]; v16bf v; };   // 32 bytes = 16 bf16 per lane

__device__ __forceinline__ unsigned short f2bf(float f) {
  unsigned u = __float_as_uint(f);
  u += 0x7FFFu + ((u >> 16) & 1u);      // round-to-nearest-even
  return (unsigned short)(u >> 16);
}

__device__ __forceinline__ unsigned lds_off(const void* p) {
  // Generic pointers to LDS carry the wave-relative LDS offset in the low
  // 32 bits (flat aperture mapping, ISA 10.2).
  return (unsigned)(size_t)p;
}

__device__ __forceinline__ void async_copy16(unsigned lds, const void* gaddr) {
  asm volatile("global_load_async_to_lds_b128 %0, %1, off"
               :: "v"(lds), "v"(gaddr) : "memory");
}

__device__ __forceinline__ void wait_async0() {
  asm volatile("s_wait_asynccnt 0x0" ::: "memory");
}

// ---------------------------------------------------------------------------
// Conversion kernels
// ---------------------------------------------------------------------------
__global__ __launch_bounds__(256)
void cvt_hidden_bf16(const float* __restrict__ src,
                     unsigned short* __restrict__ dst, int n4) {
  int i = blockIdx.x * 256 + threadIdx.x;
  if (i >= n4) return;
  float4 v = ((const float4*)src)[i];
  U64 o;
  o.x = (unsigned)f2bf(v.x) | ((unsigned)f2bf(v.y) << 16);
  o.y = (unsigned)f2bf(v.z) | ((unsigned)f2bf(v.w) << 16);
  ((U64*)dst)[i] = o;
}

// WT[k][n] = W[n][k]  (W is [E][H]) -> WT is [H][E] bf16
__global__ __launch_bounds__(256)
void cvt_w_transpose(const float* __restrict__ W, unsigned short* __restrict__ WT) {
  int i = blockIdx.x * 256 + threadIdx.x;       // over H_*E_
  int k = i / E_, n = i % E_;
  WT[i] = f2bf(W[n * H_ + k]);
}

// labBf [LP_][E_]: rows >= L zero-padded
__global__ __launch_bounds__(256)
void cvt_label_pad(const float* __restrict__ Lab, unsigned short* __restrict__ dst) {
  int i = blockIdx.x * 256 + threadIdx.x;       // over LP_*E_
  int l = i / E_;
  dst[i] = (l < L_) ? f2bf(Lab[i]) : (unsigned short)0;
}

// ---------------------------------------------------------------------------
// GEMM 1:  out1T[b][e][s] = tanh( sum_h hidden[b][s][h] * W[e][h] )   (bf16 out)
//   A = hidden_bf16 [8192][512], B = WT [512][256]
// ---------------------------------------------------------------------------
__global__ __launch_bounds__(256)
void k1_out1_tanh(const unsigned short* __restrict__ Abf,
                  const unsigned short* __restrict__ Bbf,
                  unsigned short* __restrict__ out1T) {
  __shared__ alignas(16) unsigned short sA[2][128 * 40];
  __shared__ alignas(16) unsigned short sB[2][128 * 40];
  const int t = threadIdx.x;
  const int lane = t & 31, wid = t >> 5;
  const int wm = (wid & 1) * 64, wn = (wid >> 1) * 32;
  const int m0g = blockIdx.y * 128;
  const int n0g = blockIdx.x * 128;

  v8f acc[4][2] = {};
  const int ar = t >> 2, ac = (t & 3) * 8;     // A staging: 4 thr/row
  const int bkp = t >> 4, bn0 = (t & 15) * 8;  // B staging: k-pairs

  auto stage = [&](int k0, int p) {
    #pragma unroll
    for (int rr = 0; rr < 2; ++rr) {           // A tile: async 16B/lane
      int row = ar + rr * 64;
      async_copy16(lds_off(&sA[p][row * 40 + ac]),
                   &Abf[(size_t)(m0g + row) * H_ + k0 + ac]);
    }
    const unsigned short* bp = &Bbf[(size_t)(k0 + 2 * bkp) * E_ + n0g + bn0];
    U128 r0 = *(const U128*)bp;
    U128 r1 = *(const U128*)(bp + E_);
    unsigned lo[4] = { r0.x, r0.y, r0.z, r0.w };
    unsigned hi[4] = { r1.x, r1.y, r1.z, r1.w };
    #pragma unroll
    for (int i = 0; i < 8; ++i) {
      unsigned a = (lo[i >> 1] >> ((i & 1) * 16)) & 0xFFFFu;
      unsigned b = (hi[i >> 1] >> ((i & 1) * 16)) & 0xFFFFu;
      *(unsigned*)&sB[p][(bn0 + i) * 40 + 2 * bkp] = a | (b << 16);
    }
  };

  stage(0, 0);
  const int NK = H_ / 32;
  for (int kt = 0; kt < NK; ++kt) {
    wait_async0();
    __syncthreads();
    if (kt + 1 < NK) stage((kt + 1) * 32, (kt + 1) & 1);
    const int p = kt & 1;

    FragAB fa[4], fb[2];
    const int r16 = lane & 15;
    const int koff = (lane < 16) ? 0 : 8;
    const int kb   = (lane < 16) ? 0 : 8;
    #pragma unroll
    for (int i = 0; i < 4; ++i) {
      int base = (wm + i * 16 + r16) * 40 + koff;
      fa[i].u[0] = *(const U128*)&sA[p][base];
      fa[i].u[1] = *(const U128*)&sA[p][base + 16];
    }
    #pragma unroll
    for (int j = 0; j < 2; ++j) {
      int base = (wn + j * 16 + r16) * 40 + kb;
      fb[j].u[0] = *(const U128*)&sB[p][base];
      fb[j].u[1] = *(const U128*)&sB[p][base + 16];
    }
    #pragma unroll
    for (int i = 0; i < 4; ++i)
      #pragma unroll
      for (int j = 0; j < 2; ++j)
        acc[i][j] = __builtin_amdgcn_wmma_f32_16x16x32_bf16(
            false, fa[i].v, false, fb[j].v, (short)0, acc[i][j], false, false);
  }

  const int mrow0 = (lane >> 4) * 8;
  const int ncol  = lane & 15;
  #pragma unroll
  for (int i = 0; i < 4; ++i)
    #pragma unroll
    for (int j = 0; j < 2; ++j) {
      int e = n0g + wn + j * 16 + ncol;
      #pragma unroll
      for (int r = 0; r < 8; ++r) {
        int m = m0g + wm + i * 16 + mrow0 + r;
        int b = m >> 11, s = m & 2047;
        out1T[((size_t)b * E_ + e) * S_ + s] = f2bf(tanhf(acc[i][j][r]));
      }
    }
}

// ---------------------------------------------------------------------------
// GEMM 2:  attn[b][l][s] = mask[b][s] ? sum_e lab[l][e]*out1T[b][e][s] : -1e9
//   A = labBf [8960][256], B = out1T[b] [256][2048]; f32 out (pre-softmax)
// ---------------------------------------------------------------------------
__global__ __launch_bounds__(256)
void k2_scores(const unsigned short* __restrict__ LabBf,
               const unsigned short* __restrict__ out1T,
               const int* __restrict__ mask,
               float* __restrict__ attn) {
  __shared__ alignas(16) unsigned short sA[2][128 * 40];
  __shared__ alignas(16) unsigned short sB[2][128 * 40];
  const int t = threadIdx.x;
  const int lane = t & 31, wid = t >> 5;
  const int wm = (wid & 1) * 64, wn = (wid >> 1) * 32;
  const int b = blockIdx.z;
  const int m0g = blockIdx.y * 128;
  const int n0g = blockIdx.x * 128;
  const unsigned short* Bbf = out1T + (size_t)b * E_ * S_;

  v8f acc[4][2] = {};
  const int ar = t >> 2, ac = (t & 3) * 8;
  const int bkp = t >> 4, bn0 = (t & 15) * 8;

  auto stage = [&](int k0, int p) {
    #pragma unroll
    for (int rr = 0; rr < 2; ++rr) {
      int row = ar + rr * 64;
      async_copy16(lds_off(&sA[p][row * 40 + ac]),
                   &LabBf[(size_t)(m0g + row) * E_ + k0 + ac]);
    }
    const unsigned short* bp = &Bbf[(size_t)(k0 + 2 * bkp) * S_ + n0g + bn0];
    U128 r0 = *(const U128*)bp;
    U128 r1 = *(const U128*)(bp + S_);
    unsigned lo[4] = { r0.x, r0.y, r0.z, r0.w };
    unsigned hi[4] = { r1.x, r1.y, r1.z, r1.w };
    #pragma unroll
    for (int i = 0; i < 8; ++i) {
      unsigned a  = (lo[i >> 1] >> ((i & 1) * 16)) & 0xFFFFu;
      unsigned bv = (hi[i >> 1] >> ((i & 1) * 16)) & 0xFFFFu;
      *(unsigned*)&sB[p][(bn0 + i) * 40 + 2 * bkp] = a | (bv << 16);
    }
  };

  stage(0, 0);
  const int NK = E_ / 32;
  for (int kt = 0; kt < NK; ++kt) {
    wait_async0();
    __syncthreads();
    if (kt + 1 < NK) stage((kt + 1) * 32, (kt + 1) & 1);
    const int p = kt & 1;

    FragAB fa[4], fb[2];
    const int r16 = lane & 15;
    const int koff = (lane < 16) ? 0 : 8;
    const int kb   = (lane < 16) ? 0 : 8;
    #pragma unroll
    for (int i = 0; i < 4; ++i) {
      int base = (wm + i * 16 + r16) * 40 + koff;
      fa[i].u[0] = *(const U128*)&sA[p][base];
      fa[i].u[1] = *(const U128*)&sA[p][base + 16];
    }
    #pragma unroll
    for (int j = 0; j < 2; ++j) {
      int base = (wn + j * 16 + r16) * 40 + kb;
      fb[j].u[0] = *(const U128*)&sB[p][base];
      fb[j].u[1] = *(const U128*)&sB[p][base + 16];
    }
    #pragma unroll
    for (int i = 0; i < 4; ++i)
      #pragma unroll
      for (int j = 0; j < 2; ++j)
        acc[i][j] = __builtin_amdgcn_wmma_f32_16x16x32_bf16(
            false, fa[i].v, false, fb[j].v, (short)0, acc[i][j], false, false);
  }

  const int mrow0 = (lane >> 4) * 8;
  const int ncol  = lane & 15;
  #pragma unroll
  for (int j = 0; j < 2; ++j) {
    int s = n0g + wn + j * 16 + ncol;
    int mk = mask[b * S_ + s];
    #pragma unroll
    for (int i = 0; i < 4; ++i) {
      #pragma unroll
      for (int r = 0; r < 8; ++r) {
        int l = m0g + wm + i * 16 + mrow0 + r;
        if (l < L_)
          attn[((size_t)b * L_ + l) * S_ + s] = mk ? acc[i][j][r] : NEG_INF_;
      }
    }
  }
}

// ---------------------------------------------------------------------------
// Row softmax over S (contiguous), in place on attn [B*L][2048]
// ---------------------------------------------------------------------------
__global__ __launch_bounds__(256)
void k3_softmax(float* __restrict__ attn) {
  float* p = attn + (size_t)blockIdx.x * S_;
  float4* p4 = (float4*)p;
  const int t = threadIdx.x, lane = t & 31, wid = t >> 5;
  float4 a = p4[t], c = p4[t + 256];

  float mx = fmaxf(fmaxf(fmaxf(a.x, a.y), fmaxf(a.z, a.w)),
                   fmaxf(fmaxf(c.x, c.y), fmaxf(c.z, c.w)));
  #pragma unroll
  for (int off = 16; off > 0; off >>= 1) mx = fmaxf(mx, __shfl_xor(mx, off, 32));
  __shared__ float redA[8];
  __shared__ float redB[8];
  if (lane == 0) redA[wid] = mx;
  __syncthreads();
  mx = redA[0];
  #pragma unroll
  for (int i = 1; i < 8; ++i) mx = fmaxf(mx, redA[i]);

  a.x = __expf(a.x - mx); a.y = __expf(a.y - mx);
  a.z = __expf(a.z - mx); a.w = __expf(a.w - mx);
  c.x = __expf(c.x - mx); c.y = __expf(c.y - mx);
  c.z = __expf(c.z - mx); c.w = __expf(c.w - mx);
  float sm = a.x + a.y + a.z + a.w + c.x + c.y + c.z + c.w;
  #pragma unroll
  for (int off = 16; off > 0; off >>= 1) sm += __shfl_xor(sm, off, 32);
  if (lane == 0) redB[wid] = sm;
  __syncthreads();
  float tot = redB[0];
  #pragma unroll
  for (int i = 1; i < 8; ++i) tot += redB[i];
  float inv = 1.0f / tot;

  a.x *= inv; a.y *= inv; a.z *= inv; a.w *= inv;
  c.x *= inv; c.y *= inv; c.z *= inv; c.w *= inv;
  p4[t] = a; p4[t + 256] = c;
}

// ---------------------------------------------------------------------------
// GEMM 3:  outW[b][l][h] = sum_s attn[b][l][s] * hidden[b][s][h]
//   A = attn f32 [L][2048] (bf16-converted at LDS staging), B = hiddenBf [2048][512]
// ---------------------------------------------------------------------------
__global__ __launch_bounds__(256)
void k4_weighted(const float* __restrict__ attn,
                 const unsigned short* __restrict__ hiddenBf,
                 float* __restrict__ outW) {
  __shared__ alignas(16) unsigned short sA[2][128 * 40];
  __shared__ alignas(16) unsigned short sB[2][128 * 40];
  const int t = threadIdx.x;
  const int lane = t & 31, wid = t >> 5;
  const int wm = (wid & 1) * 64, wn = (wid >> 1) * 32;
  const int b = blockIdx.z;
  const int m0g = blockIdx.y * 128;
  const int n0g = blockIdx.x * 128;
  const unsigned short* Bbf = hiddenBf + (size_t)b * S_ * H_;

  v8f acc[4][2] = {};
  const int ar = t >> 3, ac = (t & 7) * 4;     // A staging: f32, 8 thr/row
  const int bkp = t >> 4, bn0 = (t & 15) * 8;

  auto stage = [&](int k0, int p) {
    #pragma unroll
    for (int rr4 = 0; rr4 < 4; ++rr4) {        // A tile: f32 -> bf16 in VALU
      int row = ar + rr4 * 32;
      int gl = m0g + row;
      float4 v = make_float4(0.f, 0.f, 0.f, 0.f);
      if (gl < L_)
        v = *(const float4*)&attn[((size_t)b * L_ + gl) * S_ + k0 + ac];
      U64 o;
      o.x = (unsigned)f2bf(v.x) | ((unsigned)f2bf(v.y) << 16);
      o.y = (unsigned)f2bf(v.z) | ((unsigned)f2bf(v.w) << 16);
      *(U64*)&sA[p][row * 40 + ac] = o;
    }
    const unsigned short* bp = &Bbf[(size_t)(k0 + 2 * bkp) * H_ + n0g + bn0];
    U128 r0 = *(const U128*)bp;
    U128 r1 = *(const U128*)(bp + H_);
    unsigned lo[4] = { r0.x, r0.y, r0.z, r0.w };
    unsigned hi[4] = { r1.x, r1.y, r1.z, r1.w };
    #pragma unroll
    for (int i = 0; i < 8; ++i) {
      unsigned av = (lo[i >> 1] >> ((i & 1) * 16)) & 0xFFFFu;
      unsigned bv = (hi[i >> 1] >> ((i & 1) * 16)) & 0xFFFFu;
      *(unsigned*)&sB[p][(bn0 + i) * 40 + 2 * bkp] = av | (bv << 16);
    }
  };

  stage(0, 0);
  const int NK = S_ / 32;
  for (int kt = 0; kt < NK; ++kt) {
    __syncthreads();
    if (kt + 1 < NK) {
      stage((kt + 1) * 32, (kt + 1) & 1);
      if (kt + 2 < NK) {
        int k2 = (kt + 2) * 32;
        if (m0g + ar < L_)
          __builtin_prefetch(&attn[((size_t)b * L_ + m0g + ar) * S_ + k2 + ac], 0, 1);
        __builtin_prefetch(&Bbf[(size_t)(k2 + 2 * bkp) * H_ + n0g + bn0], 0, 1);
      }
    }
    const int p = kt & 1;

    FragAB fa[4], fb[2];
    const int r16 = lane & 15;
    const int koff = (lane < 16) ? 0 : 8;
    const int kb   = (lane < 16) ? 0 : 8;
    #pragma unroll
    for (int i = 0; i < 4; ++i) {
      int base = (wm + i * 16 + r16) * 40 + koff;
      fa[i].u[0] = *(const U128*)&sA[p][base];
      fa[i].u[1] = *(const U128*)&sA[p][base + 16];
    }
    #pragma unroll
    for (int j = 0; j < 2; ++j) {
      int base = (wn + j * 16 + r16) * 40 + kb;
      fb[j].u[0] = *(const U128*)&sB[p][base];
      fb[j].u[1] = *(const U128*)&sB[p][base + 16];
    }
    #pragma unroll
    for (int i = 0; i < 4; ++i)
      #pragma unroll
      for (int j = 0; j < 2; ++j)
        acc[i][j] = __builtin_amdgcn_wmma_f32_16x16x32_bf16(
            false, fa[i].v, false, fb[j].v, (short)0, acc[i][j], false, false);
  }

  const int mrow0 = (lane >> 4) * 8;
  const int ncol  = lane & 15;
  #pragma unroll
  for (int i = 0; i < 4; ++i)
    #pragma unroll
    for (int j = 0; j < 2; ++j) {
      int h = n0g + wn + j * 16 + ncol;
      #pragma unroll
      for (int r = 0; r < 8; ++r) {
        int l = m0g + wm + i * 16 + mrow0 + r;
        if (l < L_)
          outW[((size_t)b * L_ + l) * H_ + h] = acc[i][j][r];
      }
    }
}

// ---------------------------------------------------------------------------
extern "C" void kernel_launch(void* const* d_in, const int* in_sizes, int n_in,
                              void* d_out, int out_size, void* d_ws, size_t ws_size,
                              hipStream_t stream) {
  const float* hidden = (const float*)d_in[0];   // [B,S,H]
  const float* label  = (const float*)d_in[1];   // [L,E]
  const float* W      = (const float*)d_in[2];   // [E,H]
  const int*   mask   = (const int*)d_in[3];     // [B,S,1]

  float* outW = (float*)d_out;                           // [B,L,H]
  float* attn = outW + (size_t)B_ * L_ * H_;             // [B,L,S]

  char* ws = (char*)d_ws;
  unsigned short* hiddenBf = (unsigned short*)(ws);                 //  8,388,608 B
  unsigned short* out1T    = (unsigned short*)(ws + 8388608);       //  4,194,304 B
  unsigned short* WT       = (unsigned short*)(ws + 12582912);      //    262,144 B
  unsigned short* labBf    = (unsigned short*)(ws + 12845056);      //  4,587,520 B

  cvt_hidden_bf16<<<dim3(4096), dim3(256), 0, stream>>>(hidden, hiddenBf, 1048576);
  cvt_w_transpose<<<dim3(512), dim3(256), 0, stream>>>(W, WT);
  cvt_label_pad<<<dim3((LP_ * E_) / 256), dim3(256), 0, stream>>>(label, labBf);

  k1_out1_tanh<<<dim3(2, 64, 1),  dim3(256), 0, stream>>>(hiddenBf, WT, out1T);
  k2_scores   <<<dim3(16, 70, 4), dim3(256), 0, stream>>>(labBf, out1T, mask, attn);
  k3_softmax  <<<dim3(B_ * L_),   dim3(256), 0, stream>>>(attn);
  k4_weighted <<<dim3(4, 70, 4),  dim3(256), 0, stream>>>(attn, hiddenBf, outW);
}
